// MHAGRU_22127671509813
// MI455X (gfx1250) — matrix operand order repaired and checked by reading, not compiled, not run on hardware
//
#include <hip/hip_runtime.h>
#include <hip/hip_bf16.h>
#include <math.h>

// ---------------------------------------------------------------------------
// Shapes (compile-time constants from the reference)
// ---------------------------------------------------------------------------
#define B_  32
#define T_  128
#define C_  64
#define E_  8
#define H_  4
#define D_  2
#define HID_ 32
#define BT_ (B_ * T_)     // 4096
#define BC_ (B_ * C_)     // 2048

typedef __attribute__((ext_vector_type(16))) _Float16 v16h;
typedef __attribute__((ext_vector_type(8)))  float    v8f;

__device__ __forceinline__ float sigmoidf_(float x) {
    return 1.0f / (1.0f + __expf(-x));
}

// ---------------------------------------------------------------------------
// Kernel 1: xp = x @ W_in^T + b_in   (M=4096, N=64, K=64) via WMMA f16->f32
// One wave per 16x16 output tile; 2 k-chunks of 32.
// A layout (16-bit A 16x32, ISA 7.12.2): lane = g*16+row, elem i -> k=(i<8?i:i+8)+8g
// B layout: lane = g*16+col, elem i -> k=(i<8?i:i+8)+8g (same as A)
// D layout: vgpr j -> M = j + 8g, N = lane&15
// ---------------------------------------------------------------------------
__global__ void xp_wmma_kernel(const float* __restrict__ x,
                               const float* __restrict__ W_in,
                               const float* __restrict__ b_in,
                               float* __restrict__ xp) {
    const int wave = (blockIdx.x * blockDim.x + threadIdx.x) >> 5;  // 0..1023
    const int lane = threadIdx.x & 31;
    const int mt = wave >> 2;            // 256 row tiles
    const int nt = wave & 3;             // 4 col tiles
    const int m0 = mt * 16, n0 = nt * 16;
    const int rc = lane & 15;            // row (A) / col (B)
    const int g  = lane >> 4;

    v8f acc = {};
#pragma unroll
    for (int kc = 0; kc < 2; ++kc) {
        const int kb = kc * 32;
        v16h a, bm;
#pragma unroll
        for (int i = 0; i < 16; ++i) {
            const int ka = kb + ((i < 8) ? i : i + 8) + g * 8;
            a[i] = (_Float16)x[(m0 + rc) * C_ + ka];
            const int kn = kb + ((i < 8) ? i : i + 8) + g * 8;
            // xp[m,d] = sum_c x[m,c]*W_in[d,c]  ->  B[k=c][n=d] = W_in[d*64 + c]
            bm[i] = (_Float16)W_in[(n0 + rc) * C_ + kn];
        }
        acc = __builtin_amdgcn_wmma_f32_16x16x32_f16(
            false, a, false, bm, (short)0, acc, false, false);
    }
#pragma unroll
    for (int j = 0; j < 8; ++j) {
        const int m = m0 + j + g * 8;
        const int n = n0 + rc;
        xp[m * C_ + n] = acc[j] + b_in[n];
    }
}

// ---------------------------------------------------------------------------
// Kernel 2: ts_imp[b,t] = sigmoid(gelu(x @ W_ts1^T + b_ts1) @ W_ts2^T + b_ts2)
// One thread per (b,t) row; exact gelu (erf form, approximate=False).
// ---------------------------------------------------------------------------
__global__ void ts_kernel(const float* __restrict__ x,
                          const float* __restrict__ W_ts1,
                          const float* __restrict__ b_ts1,
                          const float* __restrict__ W_ts2,
                          const float* __restrict__ b_ts2,
                          float* __restrict__ ts_out) {
    const int idx = blockIdx.x * blockDim.x + threadIdx.x;   // b*T + t
    if (idx >= BT_) return;
    const float* xr = x + idx * C_;
    float acc = b_ts2[0];
#pragma unroll 4
    for (int j = 0; j < 16; ++j) {
        float s = b_ts1[j];
        for (int c = 0; c < C_; ++c) s += xr[c] * W_ts1[j * C_ + c];
        const float gl = 0.5f * s * (1.0f + erff(s * 0.70710678118654752f));
        acc += gl * W_ts2[j];
    }
    ts_out[idx] = sigmoidf_(acc);
}

// ---------------------------------------------------------------------------
// Kernel 3: per-channel GRU, 128 sequential steps.
// Block = 128 threads = 4 channels x 32 batch rows; weights staged in LDS.
// Writes hs in (B,C,T,E) layout for the attention stage.
// ---------------------------------------------------------------------------
__global__ void gru_kernel(const float* __restrict__ xp,
                           const float* __restrict__ W_ih,
                           const float* __restrict__ W_hh,
                           const float* __restrict__ b_ih,
                           const float* __restrict__ b_hh,
                           float* __restrict__ hs) {
    __shared__ float sWih[4][24], sBih[4][24], sBhh[4][24], sWhh[4][192];
    const int tid = threadIdx.x;
    const int cl  = tid >> 5;               // local channel 0..3
    const int b   = tid & 31;               // batch row
    const int c   = blockIdx.x * 4 + cl;    // global channel

    for (int i = b; i < 24; i += 32) {
        sWih[cl][i] = W_ih[c * 24 + i];
        sBih[cl][i] = b_ih[c * 24 + i];
        sBhh[cl][i] = b_hh[c * 24 + i];
    }
    for (int i = b; i < 192; i += 32) sWhh[cl][i] = W_hh[c * 192 + i];
    __syncthreads();

    float h[E_];
#pragma unroll
    for (int e = 0; e < E_; ++e) h[e] = 0.0f;

    float* hrow = hs + (size_t)(b * C_ + c) * T_ * E_;
    for (int t = 0; t < T_; ++t) {
        const float xt = xp[(b * T_ + t) * C_ + c];
        float gh[24];
#pragma unroll
        for (int gidx = 0; gidx < 24; ++gidx) {
            float s = sBhh[cl][gidx];
#pragma unroll
            for (int e = 0; e < E_; ++e) s += h[e] * sWhh[cl][gidx * E_ + e];
            gh[gidx] = s;
        }
#pragma unroll
        for (int e = 0; e < E_; ++e) {
            const float ir = xt * sWih[cl][e]      + sBih[cl][e];
            const float iz = xt * sWih[cl][8 + e]  + sBih[cl][8 + e];
            const float in = xt * sWih[cl][16 + e] + sBih[cl][16 + e];
            const float r = sigmoidf_(ir + gh[e]);
            const float z = sigmoidf_(iz + gh[8 + e]);
            const float n = tanhf(in + r * gh[16 + e]);
            h[e] = (1.0f - z) * n + z * h[e];
        }
#pragma unroll
        for (int e = 0; e < E_; ++e) hrow[t * E_ + e] = h[e];
    }
}

// ---------------------------------------------------------------------------
// Kernel 4: per-(b,c) attention, fused qkv projection + flash softmax.
// Head dim D=2 -> WMMA would be 94% padding; register-resident VALU is right.
// W_o is folded: only sum_f attn_out is needed downstream.
// Block = 128 threads (one per query row t); one block per (b,c).
// ---------------------------------------------------------------------------
__global__ void attn_kernel(const float* __restrict__ hs,
                            const float* __restrict__ W_qkv,
                            const float* __restrict__ b_qkv,
                            const float* __restrict__ W_o,
                            const float* __restrict__ b_o,
                            float* __restrict__ feat_imp,
                            float* __restrict__ tsf_imp) {
    __shared__ float q[T_][E_], k[T_][E_], v[T_][E_];
    __shared__ float red[T_];
    const int bc = blockIdx.x;          // b*C + c
    const int b  = bc >> 6;
    const int c  = bc & 63;
    const int t  = threadIdx.x;

    // qkv row projection: qkv[j] = b_qkv[j] + sum_e h[e] * W_qkv[j,e]
    const float* hr = hs + ((size_t)bc * T_ + t) * E_;
    float h[E_];
#pragma unroll
    for (int e = 0; e < E_; ++e) h[e] = hr[e];
#pragma unroll
    for (int j = 0; j < 24; ++j) {
        float s = b_qkv[j];
#pragma unroll
        for (int e = 0; e < E_; ++e) s += h[e] * W_qkv[j * E_ + e];
        if (j < 8)       q[t][j]      = s;
        else if (j < 16) k[t][j - 8]  = s;
        else             v[t][j - 16] = s;
    }
    __syncthreads();

    // flash softmax per head (online max/sum, rescaled accumulators)
    const float scale = 0.70710678118654752f;   // 1/sqrt(D), D=2
    float o[E_];
#pragma unroll
    for (int hh = 0; hh < H_; ++hh) {
        const float q0 = q[t][hh * 2], q1 = q[t][hh * 2 + 1];
        float m = -3.0e38f, l = 0.0f, o0 = 0.0f, o1 = 0.0f;
        for (int s = 0; s < T_; ++s) {
            const float sc = (q0 * k[s][hh * 2] + q1 * k[s][hh * 2 + 1]) * scale;
            const float mn = fmaxf(m, sc);
            const float corr = __expf(m - mn);
            const float p = __expf(sc - mn);
            l  = l * corr + p;
            o0 = o0 * corr + p * v[s][hh * 2];
            o1 = o1 * corr + p * v[s][hh * 2 + 1];
            m = mn;
        }
        const float rl = 1.0f / l;
        o[hh * 2]     = o0 * rl;
        o[hh * 2 + 1] = o1 * rl;
    }

    // rowsum = sum_f attn_out[b,c,t,f] = sum_e o[e] * (sum_f W_o[f,e]) + sum_f b_o[f]
    float rowsum = 0.0f;
#pragma unroll
    for (int f = 0; f < E_; ++f) rowsum += b_o[f];
#pragma unroll
    for (int e = 0; e < E_; ++e) {
        float ws = 0.0f;
#pragma unroll
        for (int f = 0; f < E_; ++f) ws += W_o[f * E_ + e];
        rowsum += o[e] * ws;
    }
    tsf_imp[(b * T_ + t) * C_ + c] = sigmoidf_(rowsum);

    red[t] = rowsum;
    __syncthreads();
    for (int st = 64; st > 0; st >>= 1) {
        if (t < st) red[t] += red[t + st];
        __syncthreads();
    }
    if (t == 0) feat_imp[bc] = sigmoidf_(red[0]);
}

// ---------------------------------------------------------------------------
// Kernel 5: out[:, -1, :] = h(t=127).reshape(B, C*E) @ W_out^T + b_out
// M=32, N=32, K=512 -> 2x2 tiles, 16 k-chunks. One block, 4 waves.
// ---------------------------------------------------------------------------
__global__ void out_wmma_kernel(const float* __restrict__ hs,
                                const float* __restrict__ W_out,
                                const float* __restrict__ b_out,
                                float* __restrict__ out) {
    const int wave = threadIdx.x >> 5;   // 0..3
    const int lane = threadIdx.x & 31;
    const int mt = wave >> 1, nt = wave & 1;
    const int m0 = mt * 16, n0 = nt * 16;
    const int rc = lane & 15;
    const int g  = lane >> 4;

    v8f acc = {};
    for (int kc = 0; kc < 16; ++kc) {
        const int kb = kc * 32;
        v16h a, bm;
#pragma unroll
        for (int i = 0; i < 16; ++i) {
            const int ka = kb + ((i < 8) ? i : i + 8) + g * 8;
            const int cidx = ka >> 3, e = ka & 7;
            // A[m][k=c*8+e] = hs[b=m][c][t=127][e]
            a[i] = (_Float16)hs[(((size_t)(m0 + rc) * C_ + cidx) * T_ + (T_ - 1)) * E_ + e];
            const int kn = kb + ((i < 8) ? i : i + 8) + g * 8;
            bm[i] = (_Float16)W_out[(n0 + rc) * (C_ * E_) + kn];
        }
        acc = __builtin_amdgcn_wmma_f32_16x16x32_f16(
            false, a, false, bm, (short)0, acc, false, false);
    }
#pragma unroll
    for (int j = 0; j < 8; ++j) {
        const int m = m0 + j + g * 8;
        const int n = n0 + rc;
        out[m * HID_ + n] = acc[j] + b_out[n];
    }
}

// ---------------------------------------------------------------------------
// Launch: d_out layout (float32, concatenated in return order):
//   [0    , 1024)  out[:, -1, :]        (32 x 32)
//   [1024 , 3072)  feat_imp             (32 x 64)
//   [3072 , 7168)  ts_imp               (32 x 128)
//   [7168 , 269312) tsf_imp             (32 x 128 x 64)
// d_ws: xp (4096*64 f32 = 1 MB) then hs (2048*128*8 f32 = 8 MB)
// ---------------------------------------------------------------------------
extern "C" void kernel_launch(void* const* d_in, const int* in_sizes, int n_in,
                              void* d_out, int out_size, void* d_ws, size_t ws_size,
                              hipStream_t stream) {
    const float* x     = (const float*)d_in[0];
    const float* W_in  = (const float*)d_in[1];
    const float* b_in  = (const float*)d_in[2];
    const float* W_ts1 = (const float*)d_in[3];
    const float* b_ts1 = (const float*)d_in[4];
    const float* W_ts2 = (const float*)d_in[5];
    const float* b_ts2 = (const float*)d_in[6];
    const float* W_ih  = (const float*)d_in[7];
    const float* W_hh  = (const float*)d_in[8];
    const float* b_ih  = (const float*)d_in[9];
    const float* b_hh  = (const float*)d_in[10];
    const float* W_qkv = (const float*)d_in[11];
    const float* b_qkv = (const float*)d_in[12];
    const float* W_o   = (const float*)d_in[13];
    const float* b_o   = (const float*)d_in[14];
    const float* W_out = (const float*)d_in[15];
    const float* b_out = (const float*)d_in[16];

    float* out      = (float*)d_out;
    float* out_last = out;                 // 1024
    float* feat_imp = out + 1024;          // 2048
    float* ts_imp   = out + 3072;          // 4096
    float* tsf_imp  = out + 7168;          // 262144

    float* xp = (float*)d_ws;              // 4096*64
    float* hs = xp + (size_t)BT_ * C_;     // 2048*128*8, (B,C,T,E)

    // 1024 tiles / 8 waves per 256-thread block = 128 blocks
    xp_wmma_kernel<<<128, 256, 0, stream>>>(x, W_in, b_in, xp);
    ts_kernel<<<BT_ / 256, 256, 0, stream>>>(x, W_ts1, b_ts1, W_ts2, b_ts2, ts_imp);
    gru_kernel<<<C_ / 4, 128, 0, stream>>>(xp, W_ih, W_hh, b_ih, b_hh, hs);
    attn_kernel<<<BC_, T_, 0, stream>>>(hs, W_qkv, b_qkv, W_o, b_o, feat_imp, tsf_imp);
    out_wmma_kernel<<<1, 128, 0, stream>>>(hs, W_out, b_out, out_last);
}
